// Score_Net_893353197948
// MI455X (gfx1250) — matrix operand order repaired and checked by reading, not compiled, not run on hardware
//
#include <hip/hip_runtime.h>
#include <math.h>

typedef _Float16 half_t;
typedef __attribute__((ext_vector_type(16))) _Float16 v16h;
typedef __attribute__((ext_vector_type(8)))  _Float16 v8h;
typedef __attribute__((ext_vector_type(8)))  float    v8f;

#define NRC 400
#define NLG 300
#define NT  700
#define NTP 704
#define KNN_K 20
#define SAMP_K 40
#define DEG_E 60
#define NE  42000        // NT * DEG_E
#define LM  1280
#define ND  256
#define ED  128
#define POSD 66
#define CDIV(a,b) (((a)+(b)-1)/(b))

__device__ __forceinline__ float silu_f(float x){ return x/(1.f+expf(-x)); }

// ---------------- generic f16 WMMA GEMM -------------------------------------
// C = act(A(MxK) @ B(KxN) + bias). One wave per 16x64 output strip (4 tiles).
// A: row-major f16, lda multiple of 32. B: fragment-packed f16:
//   Bf[(((kt*ntiles)+nt)*32 + lane)*16 + el], el -> k = kt*32+el+((lane>=16)?16:0),
//   col = nt*16 + (lane&15). K multiple of 32, N multiple of 64.
__global__ void k_gemm(const half_t* __restrict__ A, int lda,
                       const half_t* __restrict__ Bf,
                       const float*  __restrict__ bias,
                       float* __restrict__ C, int ldc,
                       int M, int N, int K, int act)
{
  int tm  = blockIdx.x;
  int tn0 = blockIdx.y * 4;
  int ntiles = N >> 4;
  int lane = threadIdx.x;                 // 0..31 (wave32)
  int row  = tm*16 + (lane & 15);
  int hiA  = (lane & 16) ? 16 : 0;
  const half_t* Arow = A + (size_t)row * lda;
  v8f acc0 = {}, acc1 = {}, acc2 = {}, acc3 = {};
  for (int k0 = 0; k0 < K; k0 += 32) {
    __builtin_prefetch(Arow + k0 + 128, 0, 0);
    v8h alo = *(const v8h*)(Arow + k0 + hiA);
    v8h ahi = *(const v8h*)(Arow + k0 + 8 + hiA);
    v16h a;
#pragma unroll
    for (int i = 0; i < 8; ++i) { a[i] = alo[i]; a[i+8] = ahi[i]; }
    const half_t* Bbase = Bf + (((size_t)(k0 >> 5) * ntiles + tn0) * 32 + lane) * 16;
    v16h b0 = *(const v16h*)(Bbase);
    v16h b1 = *(const v16h*)(Bbase + 512);
    v16h b2 = *(const v16h*)(Bbase + 1024);
    v16h b3 = *(const v16h*)(Bbase + 1536);
    acc0 = __builtin_amdgcn_wmma_f32_16x16x32_f16(false, a, false, b0, (short)0, acc0, false, false);
    acc1 = __builtin_amdgcn_wmma_f32_16x16x32_f16(false, a, false, b1, (short)0, acc1, false, false);
    acc2 = __builtin_amdgcn_wmma_f32_16x16x32_f16(false, a, false, b2, (short)0, acc2, false, false);
    acc3 = __builtin_amdgcn_wmma_f32_16x16x32_f16(false, a, false, b3, (short)0, acc3, false, false);
  }
  int mbase = tm*16 + ((lane & 16) ? 8 : 0);
#pragma unroll
  for (int q = 0; q < 4; ++q) {
    v8f acc = (q==0)?acc0:(q==1)?acc1:(q==2)?acc2:acc3;
    int n = (tn0 + q)*16 + (lane & 15);
    float bv = bias ? bias[n] : 0.f;
#pragma unroll
    for (int v = 0; v < 8; ++v) {
      int m = mbase + v;
      if (m < M) {
        float x = acc[v] + bv;
        if (act == 1) x = silu_f(x);
        C[(size_t)m * ldc + n] = x;
      }
    }
  }
}

// ---------------- small utility kernels ------------------------------------
__global__ void k_zero(float* p, int n){ int i=blockIdx.x*blockDim.x+threadIdx.x; if(i<n)p[i]=0.f; }
__global__ void k_cvt(half_t* d, const float* s, int n){ int i=blockIdx.x*blockDim.x+threadIdx.x; if(i<n)d[i]=(half_t)s[i]; }
// pack row-major KxN f32 weights into WMMA B fragment layout (zero-pad K..Kp)
__global__ void k_pack_w(half_t* d, const float* s, int K, int Kp, int N){
  int i=blockIdx.x*blockDim.x+threadIdx.x; if(i>=Kp*N)return;
  int frag = i >> 9;            // 512 halves per 32x16 fragment
  int r    = i & 511;
  int lane = r >> 4, el = r & 15;
  int ntiles = N >> 4;
  int kt = frag / ntiles, nt = frag % ntiles;
  int k  = kt*32 + el + ((lane >= 16) ? 16 : 0);
  int col= nt*16 + (lane & 15);
  d[i] = (k < K) ? (half_t)s[(size_t)k*N + col] : (half_t)0.f;
}

// ---------------- geometry --------------------------------------------------
__global__ void k_coords(const float* rec, const float* lig,
                         float* Nat, float* Ca, float* Cb, float* coord){
  int n = blockIdx.x*blockDim.x+threadIdx.x; if(n>=NT) return;
  const float* p = (n<NRC)? rec + (size_t)n*9 : lig + (size_t)(n-NRC)*9;
  float nx=p[0],ny=p[1],nz=p[2], cax=p[3],cay=p[4],caz=p[5], cx=p[6],cy=p[7],cz=p[8];
  float bx=cax-nx, by=cay-ny, bz=caz-nz;
  float ex=cx-cax, ey=cy-cay, ez=cz-caz;
  float ax=by*ez-bz*ey, ay=bz*ex-bx*ez, az=bx*ey-by*ex;
  float cbx=-0.58273431f*ax+0.56802827f*bx-0.54067466f*ex+cax;
  float cby=-0.58273431f*ay+0.56802827f*by-0.54067466f*ey+cay;
  float cbz=-0.58273431f*az+0.56802827f*bz-0.54067466f*ez+caz;
  Nat[n*3]=nx; Nat[n*3+1]=ny; Nat[n*3+2]=nz;
  Ca[n*3]=cax; Ca[n*3+1]=cay; Ca[n*3+2]=caz;
  Cb[n*3]=cbx; Cb[n*3+1]=cby; Cb[n*3+2]=cbz;
  coord[n*3]=cax; coord[n*3+1]=cay; coord[n*3+2]=caz;
}

__device__ __forceinline__ unsigned wanghash(unsigned s){
  s=(s^61u)^(s>>16); s*=9u; s^=s>>4; s*=0x27d4eb2du; s^=s>>15; return s;
}

// one block (256 threads) per node: KNN (min-dist) + 40 Gumbel-top-k samples
__global__ void k_graph(const float* __restrict__ Ca, int* __restrict__ srcB, int* __restrict__ dstB){
  int i = blockIdx.x; int t = threadIdx.x;
  __shared__ float d[NT];
  __shared__ float sc[NT];
  __shared__ unsigned char taken[NT];
  __shared__ float redv[256];
  __shared__ int   redi[256];
  float cx=Ca[i*3], cy=Ca[i*3+1], cz=Ca[i*3+2];
  for(int j=t;j<NT;j+=256){
    float dx=cx-Ca[j*3], dy=cy-Ca[j*3+1], dz=cz-Ca[j*3+2];
    d[j]=sqrtf(fmaxf(dx*dx+dy*dy+dz*dz,1e-12f));
    taken[j]=0;
  }
  __syncthreads();
  for(int s=0;s<KNN_K;s++){
    float bv=3e38f; int bi=0x7fffffff;
    for(int j=t;j<NT;j+=256)
      if(!taken[j]){ float dv=d[j]; if(dv<bv||(dv==bv&&j<bi)){bv=dv;bi=j;} }
    redv[t]=bv; redi[t]=bi; __syncthreads();
    for(int o=128;o>0;o>>=1){
      if(t<o){
        if(redv[t+o]<redv[t]||(redv[t+o]==redv[t]&&redi[t+o]<redi[t])){
          redv[t]=redv[t+o]; redi[t]=redi[t+o];
        }
      }
      __syncthreads();
    }
    if(t==0){ int j=redi[0]; taken[j]=1; dstB[i*DEG_E+s]=j; srcB[i*DEG_E+s]=i; }
    __syncthreads();
  }
  for(int j=t;j<NT;j+=256){
    unsigned h = wanghash((unsigned)(i*1024+j)*2654435761u + 12345u);
    float u = (h>>8)*(1.f/16777216.f);
    float g = -logf(-logf(u+1e-12f)+1e-12f);
    sc[j] = -3.f*logf(fmaxf(d[j],1e-10f)) + g;
  }
  __syncthreads();
  for(int s=0;s<SAMP_K;s++){
    float bv=-3e38f; int bi=0x7fffffff;
    for(int j=t;j<NT;j+=256)
      if(!taken[j]){ float v=sc[j]; if(v>bv||(v==bv&&j<bi)){bv=v;bi=j;} }
    redv[t]=bv; redi[t]=bi; __syncthreads();
    for(int o=128;o>0;o>>=1){
      if(t<o){
        if(redv[t+o]>redv[t]||(redv[t+o]==redv[t]&&redi[t+o]<redi[t])){
          redv[t]=redv[t+o]; redi[t]=redi[t+o];
        }
      }
      __syncthreads();
    }
    if(t==0){ int j=redi[0]; taken[j]=1; dstB[i*DEG_E+KNN_K+s]=j; srcB[i*DEG_E+KNN_K+s]=i; }
    __syncthreads();
  }
}

__device__ __forceinline__ float dihedralf(const float* p0,const float* p1,const float* p2,const float* p3){
  float b0x=p0[0]-p1[0], b0y=p0[1]-p1[1], b0z=p0[2]-p1[2];
  float b1x=p2[0]-p1[0], b1y=p2[1]-p1[1], b1z=p2[2]-p1[2];
  float b2x=p3[0]-p2[0], b2y=p3[1]-p2[1], b2z=p3[2]-p2[2];
  float nb=sqrtf(b1x*b1x+b1y*b1y+b1z*b1z)+1e-8f;
  b1x/=nb; b1y/=nb; b1z/=nb;
  float d0=b0x*b1x+b0y*b1y+b0z*b1z;
  float vx=b0x-d0*b1x, vy=b0y-d0*b1y, vz=b0z-d0*b1z;
  float d2=b2x*b1x+b2y*b1y+b2z*b1z;
  float wx=b2x-d2*b1x, wy=b2y-d2*b1y, wz=b2z-d2*b1z;
  float x=vx*wx+vy*wy+vz*wz;
  float cxx=b1y*vz-b1z*vy, cxy=b1z*vx-b1x*vz, cxz=b1x*vy-b1y*vx;
  float y=cxx*wx+cxy*wy+cxz*wz;
  return atan2f(y,x)*57.29577951308232f;
}

__device__ __forceinline__ int get_binf(float x, float lo, float hi, int nb){
  float step=(hi-lo)/(float)(nb-2);
  int c=0;
  for(int k=0;k<nb-1;k++) c += (x > lo + (float)k*step) ? 1 : 0;
  return c;
}

__global__ void k_edge_geom(const int* src, const int* dst,
                            const float* Nat, const float* Ca, const float* Cb,
                            float* cd, float* radial, int* bins){
  int e = blockIdx.x*blockDim.x+threadIdx.x; if(e>=NE)return;
  int i=src[e], j=dst[e];
  float cai[3]={Ca[i*3],Ca[i*3+1],Ca[i*3+2]};
  float caj[3]={Ca[j*3],Ca[j*3+1],Ca[j*3+2]};
  float cbi[3]={Cb[i*3],Cb[i*3+1],Cb[i*3+2]};
  float cbj[3]={Cb[j*3],Cb[j*3+1],Cb[j*3+2]};
  float ni[3]={Nat[i*3],Nat[i*3+1],Nat[i*3+2]};
  float dx=cai[0]-caj[0], dy=cai[1]-caj[1], dz=cai[2]-caj[2];
  cd[e*3]=dx; cd[e*3+1]=dy; cd[e*3+2]=dz;
  radial[e]=dx*dx+dy*dy+dz*dz;
  float bx=cbi[0]-cbj[0], by=cbi[1]-cbj[1], bz=cbi[2]-cbj[2];
  float db=sqrtf(bx*bx+by*by+bz*bz+1e-12f);
  int keep = (db<22.f) && (i!=j);
  float omega=dihedralf(cai,cbi,cbj,caj);
  float theta=dihedralf(ni,cai,cbi,cbj);
  float vx=cai[0]-cbi[0], vy=cai[1]-cbi[1], vz=cai[2]-cbi[2];
  float wx=cbj[0]-cbi[0], wy=cbj[1]-cbi[1], wz=cbj[2]-cbi[2];
  float nv=sqrtf(vx*vx+vy*vy+vz*vz), nw=sqrtf(wx*wx+wy*wy+wz*wz);
  float cosphi=(vx*wx+vy*wy+vz*wz)/(nv*nw+1e-8f);
  cosphi=fminf(1.f,fmaxf(-1.f,cosphi));
  float phi=acosf(cosphi)*57.29577951308232f;
  bins[e*4+0]=get_binf(db,3.25f,50.75f,40);
  bins[e*4+1]=keep?get_binf(omega,-180.f,180.f,24):0;
  bins[e*4+2]=keep?get_binf(theta,-180.f,180.f,24):0;
  bins[e*4+3]=keep?get_binf(phi,0.f,180.f,12):0;
}

// ---------------- packing kernels ------------------------------------------
__global__ void k_pack_x(half_t* x16, const float* rec, const float* lig){
  long i=(long)blockIdx.x*blockDim.x+threadIdx.x; if(i>=(long)NTP*LM)return;
  int n=(int)(i/LM), c=(int)(i%LM);
  float v = (n<NRC)?rec[(size_t)n*LM+c] : (n<NT)?lig[(size_t)(n-NRC)*LM+c] : 0.f;
  x16[i]=(half_t)v;
}
__global__ void k_pack_pm(half_t* A, const int* src, const int* dst, const float* pm){
  long i=(long)blockIdx.x*blockDim.x+threadIdx.x; if(i>=(long)NE*96)return;
  int e=(int)(i/96), c=(int)(i%96);
  float v = (c<POSD)? pm[((size_t)src[e]*NT + dst[e])*POSD + c] : 0.f;
  A[i]=(half_t)v;
}
__global__ void k_edge_attr(float* eaF, half_t* ea16, const int* bins, const float* spat){
  long i=(long)blockIdx.x*blockDim.x+threadIdx.x; if(i>=(long)NE*ED)return;
  int e=(int)(i/ED), c=(int)(i%ED);
  int b0=bins[e*4],b1=bins[e*4+1],b2=bins[e*4+2],b3=bins[e*4+3];
  float v=eaF[i];
  v += spat[b0*ED+c] + spat[(40+b1)*ED+c] + spat[(64+b2)*ED+c] + spat[(88+b3)*ED+c];
  eaF[i]=v; ea16[i]=(half_t)v;
}
__global__ void k_pack_e1(half_t* A, const half_t* h16, const int* src, const int* dst,
                          const float* radial, const half_t* ea16){
  long i=(long)blockIdx.x*blockDim.x+threadIdx.x; if(i>=(long)NE*672)return;
  int e=(int)(i/672), c=(int)(i%672);
  half_t v;
  if(c<256)       v = h16[(size_t)src[e]*ND + c];
  else if(c<512)  v = h16[(size_t)dst[e]*ND + (c-256)];
  else if(c==512) v = (half_t)radial[e];
  else if(c<641)  v = ea16[(size_t)e*ED + (c-513)];
  else            v = (half_t)0.f;
  A[i]=v;
}
__global__ void k_pack_n1(half_t* A, const float* hF, const float* agg){
  long i=(long)blockIdx.x*blockDim.x+threadIdx.x; if(i>=(long)NTP*512)return;
  int n=(int)(i/512), c=(int)(i%512);
  float v = (c<ND)? hF[(size_t)n*ND+c] : agg[(size_t)n*ND+(c-ND)];
  A[i]=(half_t)v;
}

// ---------------- EGCL element-wise pieces ----------------------------------
__global__ void k_att(float* m, half_t* m16, const float* attw, const float* attb){
  int e=blockIdx.x*blockDim.x+threadIdx.x; if(e>=NE)return;
  const float* row=m+(size_t)e*ND;
  float a=attb[0];
  for(int c=0;c<ND;c++) a += attw[c]*row[c];
  float s=1.f/(1.f+expf(-a));
  float* wr=m+(size_t)e*ND; half_t* hr=m16+(size_t)e*ND;
  for(int c=0;c<ND;c++){ float v=wr[c]*s; wr[c]=v; hr[c]=(half_t)v; }
}
__global__ void k_agg(float* agg, const float* m){
  int n=blockIdx.x; int c=threadIdx.x;
  float s=0.f;
  if(n<NT){ for(int e=0;e<DEG_E;e++) s += m[((size_t)n*DEG_E+e)*ND + c]; }
  agg[(size_t)n*ND+c]=s;
}
__global__ void k_h_update(float* hF, half_t* h16, const float* delta){
  long i=(long)blockIdx.x*blockDim.x+threadIdx.x; if(i>=(long)NT*ND)return;
  float v=hF[i]+delta[i]; hF[i]=v; h16[i]=(half_t)v;
}
__global__ void k_trans(float* trans, const float* w1, const float* c2w,
                        const float* cd){
  int e=blockIdx.x*blockDim.x+threadIdx.x; if(e>=NE)return;
  const float* row=w1+(size_t)e*ND;
  float w=0.f;
  for(int c=0;c<ND;c++) w += c2w[c]*row[c];
  for(int k=0;k<3;k++){
    float tcl=cd[e*3+k]*w;
    tcl=fminf(2.f,fmaxf(-2.f,tcl));
    trans[e*3+k]=tcl;
  }
}
__global__ void k_coord_apply(float* coord, const float* trans){
  int n=blockIdx.x*blockDim.x+threadIdx.x; if(n>=NT)return;
  float mask=(n>=NRC)?1.f:0.f;
  float s0=0,s1=0,s2=0;
  for(int e=0;e<DEG_E;e++){
    s0+=trans[((size_t)n*DEG_E+e)*3+0];
    s1+=trans[((size_t)n*DEG_E+e)*3+1];
    s2+=trans[((size_t)n*DEG_E+e)*3+2];
  }
  coord[n*3+0]+=s0/60.f*mask; coord[n*3+1]+=s1/60.f*mask; coord[n*3+2]+=s2/60.f*mask;
}

// ---------------- heads -----------------------------------------------------
__global__ void k_ires3(float* out, const float* z2, const float* w, const float* b){
  int n=blockIdx.x*blockDim.x+threadIdx.x; if(n>=NT)return;
  const float* row=z2+(size_t)n*512;
  float a=b[0];
  for(int k=0;k<512;k++) a += w[k]*row[k];
  out[7+n]=a;
}
__global__ void k_energy(float* acc, const float* pairR, const float* pairL,
                         const float* rec, const float* lig,
                         const float* g, const float* bb, const float* en2w){
  int pid=blockIdx.x*blockDim.x+threadIdx.x; if(pid>=NRC*NLG)return;
  int i=pid/NLG, j=pid%NLG;
  float dx=rec[i*9+3]-lig[j*9+3], dy=rec[i*9+4]-lig[j*9+4], dz=rec[i*9+5]-lig[j*9+5];
  float D=sqrtf(dx*dx+dy*dy+dz*dz+1e-12f);
  if(D>=20.f) return;
  const float* pr=pairR+(size_t)i*ND; const float* pl=pairL+(size_t)j*ND;
  float sm=0.f, sq=0.f;
  for(int c=0;c<ND;c++){ float x=pr[c]+pl[c]; sm+=x; sq+=x*x; }
  float m=sm*(1.f/ND);
  float var=fmaxf(sq*(1.f/ND)-m*m,0.f);
  float inv=1.f/sqrtf(var+1e-5f);
  float a=0.f;
  for(int c=0;c<ND;c++){
    float x=pr[c]+pl[c];
    float y=(x-m)*inv*g[c]+bb[c];
    a += silu_f(y)*en2w[c];
  }
  atomicAdd(&acc[0], a);
  atomicAdd(&acc[1], 1.f);
}

__device__ float blk_sum(float v, float* red, int t){
  red[t]=v; __syncthreads();
  for(int o=128;o>0;o>>=1){ if(t<o) red[t]+=red[t+o]; __syncthreads(); }
  float r=red[0]; __syncthreads(); return r;
}
__device__ float scale_head(const float* w1,const float* g,const float* b,const float* w2,
                            float nrm,const float* temb,float* red,int t){
  float s = nrm*w1[t];
  for(int k=0;k<256;k++) s += temb[k]*w1[(1+k)*256+t];
  float m = blk_sum(s,red,t)*(1.f/256.f);
  float d = s-m;
  float v = blk_sum(d*d,red,t)*(1.f/256.f);
  float y = d/sqrtf(v+1e-5f)*g[t]+b[t];
  y = silu_f(y);
  float dv = blk_sum(y*w2[t],red,t);
  return dv>30.f ? dv : log1pf(expf(dv));
}
__global__ void k_final(float* out, const float* coord, const float* lig,
                        const float* tptr, const float* tW, const float* tlin,
                        const float* tr1w, const float* trg, const float* trb, const float* tr2w,
                        const float* ro1w, const float* rog, const float* rob, const float* ro2w,
                        const float* acc){
  __shared__ float red[256];
  __shared__ float fv[256];
  __shared__ float temb[256];
  __shared__ float res[6];
  int t=threadIdx.x;
  float lf0=0,lf1=0,lf2=0,lc0=0,lc1=0,lc2=0;
  for(int j=t;j<NLG;j+=256){
    float rx=lig[j*9+3],ry=lig[j*9+4],rz=lig[j*9+5];
    float fx=coord[(NRC+j)*3+0]-rx, fy=coord[(NRC+j)*3+1]-ry, fz=coord[(NRC+j)*3+2]-rz;
    lf0+=fx; lf1+=fy; lf2+=fz;
    lc0+=ry*fz-rz*fy; lc1+=rz*fx-rx*fz; lc2+=rx*fy-ry*fx;
  }
  float s;
  s=blk_sum(lf0,red,t); if(t==0)res[0]=s/(float)NLG;
  s=blk_sum(lf1,red,t); if(t==0)res[1]=s/(float)NLG;
  s=blk_sum(lf2,red,t); if(t==0)res[2]=s/(float)NLG;
  s=blk_sum(lc0,red,t); if(t==0)res[3]=s/(float)NLG;
  s=blk_sum(lc1,red,t); if(t==0)res[4]=s/(float)NLG;
  s=blk_sum(lc2,red,t); if(t==0)res[5]=s/(float)NLG;
  __syncthreads();
  float trp0=res[0],trp1=res[1],trp2=res[2], rop0=res[3],rop1=res[4],rop2=res[5];
  float tv=tptr[0];
  if(t<128) fv[t]=sinf(tv*tW[t]*6.283185307f);
  else      fv[t]=cosf(tv*tW[t-128]*6.283185307f);
  __syncthreads();
  { float a=0.f; for(int k=0;k<256;k++) a+=fv[k]*tlin[k*256+t]; temb[t]=1.f/(1.f+expf(-a)); }
  __syncthreads();
  float trn=sqrtf(trp0*trp0+trp1*trp1+trp2*trp2+1e-12f);
  float str=scale_head(tr1w,trg,trb,tr2w,trn,temb,red,t);
  float ron=sqrtf(rop0*rop0+rop1*rop1+rop2*rop2+1e-12f);
  float sro=scale_head(ro1w,rog,rob,ro2w,ron,temb,red,t);
  if(t==0){
    out[0]=trp0/(trn+1e-6f)*str; out[1]=trp1/(trn+1e-6f)*str; out[2]=trp2/(trn+1e-6f)*str;
    out[3]=rop0/(ron+1e-6f)*sro; out[4]=rop1/(ron+1e-6f)*sro; out[5]=rop2/(ron+1e-6f)*sro;
    out[6]=acc[0]/(acc[1]+1e-6f);
  }
}

// ---------------- host ------------------------------------------------------
static void gemm(hipStream_t s, const half_t* A,int lda,const half_t* Bf,
                 const float* bias,float* C,int ldc,int M,int N,int K,int act){
  dim3 g(CDIV(M,16), N/64);
  k_gemm<<<g,32,0,s>>>(A,lda,Bf,bias,C,ldc,M,N,K,act);
}

extern "C" void kernel_launch(void* const* d_in, const int* in_sizes, int n_in,
                              void* d_out, int out_size, void* d_ws, size_t ws_size,
                              hipStream_t stream) {
  (void)in_sizes; (void)n_in; (void)out_size; (void)ws_size;
  const float* rec_x   = (const float*)d_in[0];
  const float* lig_x   = (const float*)d_in[1];
  const float* rec_pos = (const float*)d_in[2];
  const float* lig_pos = (const float*)d_in[3];
  const float* t_in    = (const float*)d_in[4];
  const float* pm      = (const float*)d_in[5];
  auto PP=[&](int i)->const float*{ return (const float*)d_in[6+i]; };
  // params leaves in jax tree-flatten (sorted-key) order
  const float* en1_w=PP(0); const float* en2_w=PP(1);
  const float* en_ln_b=PP(2); const float* en_ln_g=PP(3);
  const float* ires1_b=PP(4); const float* ires1_w=PP(5);
  const float* ires2_b=PP(6); const float* ires2_w=PP(7);
  const float* ires3_b=PP(8); const float* ires3_w=PP(9);
  // layers: base 10 + 13*L : att_b,att_w,c1_b,c1_w,c2_w,e1_b,e1_w,e2_b,e2_w,n1_b,n1_w,n2_b,n2_w
  const float* pos_emb=PP(62);
  const float* rot1_w=PP(63); const float* rot2_w=PP(64);
  const float* rot_ln_b=PP(65); const float* rot_ln_g=PP(66);
  const float* single_emb=PP(67); const float* spat_emb=PP(68);
  const float* t_W=PP(69); const float* t_lin=PP(70);
  const float* tr1_w=PP(71); const float* tr2_w=PP(72);
  const float* tr_ln_b=PP(73); const float* tr_ln_g=PP(74);

  // workspace bump allocator
  size_t off=0;
  auto alloc=[&](size_t bytes)->char*{
    size_t o=off; off=(off+bytes+255)&~(size_t)255; return (char*)d_ws+o;
  };
  float* Nat   =(float*)alloc((size_t)NT*3*4);
  float* Ca    =(float*)alloc((size_t)NT*3*4);
  float* Cb    =(float*)alloc((size_t)NT*3*4);
  float* coord =(float*)alloc((size_t)NTP*3*4);
  int*   srcB  =(int*)  alloc((size_t)NE*4);
  int*   dstB  =(int*)  alloc((size_t)NE*4);
  float* cd    =(float*)alloc((size_t)NE*3*4);
  float* radial=(float*)alloc((size_t)NE*4);
  int*   bins  =(int*)  alloc((size_t)NE*4*4);
  float* acc   =(float*)alloc(256);
  half_t* x16  =(half_t*)alloc((size_t)NTP*LM*2);
  half_t* Wemb =(half_t*)alloc((size_t)LM*ND*2);
  half_t* Wpm  =(half_t*)alloc((size_t)96*ED*2);
  half_t* We1  =(half_t*)alloc((size_t)672*ND*2);
  half_t* We2  =(half_t*)alloc((size_t)ND*ND*2);
  half_t* Wn1  =(half_t*)alloc((size_t)512*ND*2);
  half_t* Wn2  =(half_t*)alloc((size_t)ND*ND*2);
  half_t* Wc1  =(half_t*)alloc((size_t)ND*ND*2);
  half_t* Wi1  =(half_t*)alloc((size_t)ND*512*2);
  half_t* Wi2  =(half_t*)alloc((size_t)512*512*2);
  half_t* Wen1 =(half_t*)alloc((size_t)512*ND*2);   // two 256x256 fragments
  float* hF    =(float*)alloc((size_t)NTP*ND*4);
  half_t* h16  =(half_t*)alloc((size_t)NTP*ND*2);
  float* agg   =(float*)alloc((size_t)NTP*ND*4);
  half_t* nA16 =(half_t*)alloc((size_t)NTP*512*2); // n1-input / u16 / z1f16
  float* nodeF =(float*)alloc((size_t)NTP*512*4);  // u / delta / z1 / z2
  float* pairR =(float*)alloc((size_t)NRC*ND*4);
  float* pairL =(float*)alloc((size_t)304*ND*4);
  float* eaF   =(float*)alloc((size_t)NE*ED*4);
  half_t* ea16 =(half_t*)alloc((size_t)NE*ED*2);
  half_t* Ae1  =(half_t*)alloc((size_t)NE*672*2);  // also reused as pm-A (NE x 96)
  float* mF    =(float*)alloc((size_t)NE*ND*4);
  half_t* m16  =(half_t*)alloc((size_t)NE*ND*2);
  float* trans =(float*)alloc((size_t)NE*3*4);
  float* outF  =(float*)d_out;

  // ---- geometry & graph ----
  k_zero<<<1,64,0,stream>>>(acc,64);
  k_coords<<<CDIV(NT,256),256,0,stream>>>(rec_pos,lig_pos,Nat,Ca,Cb,coord);
  k_graph<<<NT,256,0,stream>>>(Ca,srcB,dstB);
  k_edge_geom<<<CDIV(NE,256),256,0,stream>>>(srcB,dstB,Nat,Ca,Cb,cd,radial,bins);

  // ---- edge attr: positional GEMM + spatial embedding gathers ----
  k_pack_w<<<CDIV(96*ED,256),256,0,stream>>>(Wpm,pos_emb,POSD,96,ED);
  k_pack_pm<<<CDIV((long)NE*96,256),256,0,stream>>>(Ae1,srcB,dstB,pm);
  gemm(stream,Ae1,96,Wpm,nullptr,eaF,ED,NE,ED,96,0);
  k_edge_attr<<<CDIV((long)NE*ED,256),256,0,stream>>>(eaF,ea16,bins,spat_emb);

  // ---- node embedding ----
  k_pack_w<<<CDIV(LM*ND,256),256,0,stream>>>(Wemb,single_emb,LM,LM,ND);
  k_pack_x<<<CDIV((long)NTP*LM,256),256,0,stream>>>(x16,rec_x,lig_x);
  gemm(stream,x16,LM,Wemb,nullptr,hF,ND,NTP,ND,LM,0);
  k_cvt<<<CDIV(NTP*ND,256),256,0,stream>>>(h16,hF,NTP*ND);

  // ---- EGCL layers ----
  for(int L=0;L<4;L++){
    int b0=10+13*L;
    const float* att_b=PP(b0+0); const float* att_w=PP(b0+1);
    const float* c1_b =PP(b0+2); const float* c1_w =PP(b0+3);
    const float* c2_w =PP(b0+4);
    const float* e1_b =PP(b0+5); const float* e1_w =PP(b0+6);
    const float* e2_b =PP(b0+7); const float* e2_w =PP(b0+8);
    const float* n1_b =PP(b0+9); const float* n1_w =PP(b0+10);
    const float* n2_b =PP(b0+11); const float* n2_w=PP(b0+12);
    k_pack_w<<<CDIV(672*ND,256),256,0,stream>>>(We1,e1_w,641,672,ND);
    k_pack_w<<<CDIV(ND*ND,256),256,0,stream>>>(We2,e2_w,ND,ND,ND);
    k_pack_w<<<CDIV(512*ND,256),256,0,stream>>>(Wn1,n1_w,512,512,ND);
    k_pack_w<<<CDIV(ND*ND,256),256,0,stream>>>(Wn2,n2_w,ND,ND,ND);
    if(L==3) k_pack_w<<<CDIV(ND*ND,256),256,0,stream>>>(Wc1,c1_w,ND,ND,ND);

    k_pack_e1<<<CDIV((long)NE*672,256),256,0,stream>>>(Ae1,h16,srcB,dstB,radial,ea16);
    gemm(stream,Ae1,672,We1,e1_b,mF,ND,NE,ND,672,1);           // m1 = silu(e1)
    k_cvt<<<CDIV((long)NE*ND,256),256,0,stream>>>(m16,mF,NE*ND);
    gemm(stream,m16,ND,We2,e2_b,mF,ND,NE,ND,ND,1);             // m2 = silu(e2)
    k_att<<<CDIV(NE,256),256,0,stream>>>(mF,m16,att_w,att_b);  // m = m2*sigmoid(att)
    k_agg<<<NTP,ND,0,stream>>>(agg,mF);
    k_pack_n1<<<CDIV((long)NTP*512,256),256,0,stream>>>(nA16,hF,agg);
    gemm(stream,nA16,512,Wn1,n1_b,nodeF,ND,NTP,ND,512,1);      // u = silu(n1)
    k_cvt<<<CDIV(NTP*ND,256),256,0,stream>>>(nA16,nodeF,NTP*ND);
    gemm(stream,nA16,ND,Wn2,n2_b,nodeF,ND,NTP,ND,ND,0);        // delta = n2(u)
    k_h_update<<<CDIV(NT*ND,256),256,0,stream>>>(hF,h16,nodeF);
    if(L==3){
      gemm(stream,m16,ND,Wc1,c1_b,mF,ND,NE,ND,ND,1);           // silu(c1(m))
      k_trans<<<CDIV(NE,256),256,0,stream>>>(trans,mF,c2_w,cd);
      k_coord_apply<<<CDIV(NT,256),256,0,stream>>>(coord,trans);
    }
  }

  // ---- ires head ----
  k_pack_w<<<CDIV(ND*512,256),256,0,stream>>>(Wi1,ires1_w,ND,ND,512);
  k_pack_w<<<CDIV(512*512,256),256,0,stream>>>(Wi2,ires2_w,512,512,512);
  gemm(stream,h16,ND,Wi1,ires1_b,nodeF,512,NTP,512,ND,1);      // z1
  k_cvt<<<CDIV(NTP*512,256),256,0,stream>>>(nA16,nodeF,NTP*512);
  gemm(stream,nA16,512,Wi2,ires2_b,nodeF,512,NTP,512,512,1);   // z2
  k_ires3<<<CDIV(NT,256),256,0,stream>>>(outF,nodeF,ires3_w,ires3_b);

  // ---- energy ----
  k_pack_w<<<CDIV(ND*ND,256),256,0,stream>>>(Wen1,en1_w,ND,ND,ND);
  k_pack_w<<<CDIV(ND*ND,256),256,0,stream>>>(Wen1+(size_t)ND*ND,en1_w+(size_t)ND*ND,ND,ND,ND);
  gemm(stream,h16,ND,Wen1,nullptr,pairR,ND,NRC,ND,ND,0);
  gemm(stream,h16+(size_t)NRC*ND,ND,Wen1+(size_t)ND*ND,nullptr,pairL,ND,NLG,ND,ND,0);
  k_energy<<<CDIV(NRC*NLG,256),256,0,stream>>>(acc,pairR,pairL,rec_pos,lig_pos,
                                               en_ln_g,en_ln_b,en2_w);

  // ---- tr / rot scores + write out[0..6] ----
  k_final<<<1,256,0,stream>>>(outF,coord,lig_pos,t_in,t_W,t_lin,
                              tr1_w,tr_ln_g,tr_ln_b,tr2_w,
                              rot1_w,rot_ln_g,rot_ln_b,rot2_w,acc);
}
